// LSTM_7481833029848
// MI455X (gfx1250) — matrix operand order repaired and checked by reading, not compiled, not run on hardware
//
#include <hip/hip_runtime.h>
#include <hip/hip_bf16.h>

// ---------------------------------------------------------------------------
// LSTM: B=256 batch, S=256 steps, H=1024 hidden, C=10 classes.
// Per step: fused 4-gate GEMM (bf16 WMMA, f32 accum) + gate nonlinearity +
// cell/hidden update, all in one kernel. 256 sequential launches in the graph.
//
// Wave-level tiling: each wave owns a 32(batch) x 16(hidden) patch for ALL
// FOUR gates -> 8 accumulators, 2 A fragments, 4 B fragments per K-step.
// Each B fragment feeds two WMMAs (two M-subtiles), cutting per-WMMA load
// traffic from 40 B/lane to 24 B/lane vs. one-patch-per-wave tiling.
// ---------------------------------------------------------------------------

static constexpr int kB = 256;
static constexpr int kS = 256;
static constexpr int kH = 1024;
static constexpr int kC = 10;

typedef __attribute__((ext_vector_type(16))) __bf16 v16bf;
typedef __attribute__((ext_vector_type(8)))  __bf16 v8bf;
typedef __attribute__((ext_vector_type(8)))  float  v8f;

// ---------------------------------------------------------------------------
// fp32 -> bf16 weight conversion (once per launch; weights then live in L2)
// ---------------------------------------------------------------------------
__global__ __launch_bounds__(256)
void convert_weights(const float* __restrict__ g, const float* __restrict__ i,
                     const float* __restrict__ f, const float* __restrict__ o,
                     __bf16* __restrict__ wg, __bf16* __restrict__ wi,
                     __bf16* __restrict__ wf, __bf16* __restrict__ wo, int n) {
    int idx = blockIdx.x * blockDim.x + threadIdx.x;
    if (idx < n) {
        wg[idx] = (__bf16)g[idx];
        wi[idx] = (__bf16)i[idx];
        wf[idx] = (__bf16)f[idx];
        wo[idx] = (__bf16)o[idx];
    }
}

__global__ __launch_bounds__(256)
void init_state(__bf16* __restrict__ h0, float* __restrict__ c, int n) {
    int idx = blockIdx.x * blockDim.x + threadIdx.x;
    if (idx < n) {
        h0[idx] = (__bf16)0.0f;
        c[idx]  = 0.0f;
    }
}

__device__ __forceinline__ float sigmoidf_fast(float v) {
    return 1.0f / (1.0f + __expf(-v));
}

// ---------------------------------------------------------------------------
// One LSTM timestep.
// Grid: (kB/32, kH/64) = (8,16) = 128 WGs. Block: 128 threads = 4 waves.
// Wave w: rows m0 = bx*32 .. +31 (two 16-row M-subtiles),
//         cols n0 = by*64 + w*16 .. +15, all four gates.
// ---------------------------------------------------------------------------
__global__ __launch_bounds__(128)
void lstm_step(const __bf16* __restrict__ hin,   // [B,H] bf16 (WMMA A)
               __bf16*       __restrict__ hout,  // [B,H] bf16 (next step A)
               float*        __restrict__ cbuf,  // [B,H] f32 cell state
               float*        __restrict__ hf32,  // [B,H] f32 hidden (for head)
               const __bf16* __restrict__ Wg, const __bf16* __restrict__ Wi,
               const __bf16* __restrict__ Wf, const __bf16* __restrict__ Wo,
               const float*  __restrict__ x,     // [B,S] f32
               const float*  __restrict__ fgx, const float* __restrict__ fixv,
               const float*  __restrict__ ffx, const float* __restrict__ foxv,
               const float*  __restrict__ bg,   // NOTE: reference uses b_g for all 4 gates
               int t) {
    const int lane = threadIdx.x & 31;
    const int wave = threadIdx.x >> 5;            // 0..3
    const int m0 = blockIdx.x * 32;
    const int n0 = blockIdx.y * 64 + wave * 16;

    const int lmod = lane & 15;   // sub-tile row (A) / column (B,C,D)
    const int lhi  = lane >> 4;   // K-half selector

    // A fragments (contiguous K order chosen here, same as B):
    //  lanes 0-15 : row m+lane,    K = {k0+0..15}
    //  lanes 16-31: row m+lane-16, K = {k0+16..31}
    const __bf16* aRow0 = hin + (size_t)(m0 + lmod) * kH + lhi * 16;       // rows m0..m0+15
    const __bf16* aRow1 = aRow0 + (size_t)16 * kH;                          // rows m0+16..m0+31

    // B fragments (same contiguous K order as A): lane holds column n0+lmod,
    //  K = k0 + lhi*16 + (0..15), contiguous since W is row-major [j,k].
    const size_t bOff = (size_t)(n0 + lmod) * kH + lhi * 16;
    const __bf16* bg_p = Wg + bOff;
    const __bf16* bi_p = Wi + bOff;
    const __bf16* bf_p = Wf + bOff;
    const __bf16* bo_p = Wo + bOff;

    // acc[mg][gate]: mg = M-subtile (0: rows m0.., 1: rows m0+16..)
    v8f acc[2][4];
    #pragma unroll
    for (int mg = 0; mg < 2; ++mg)
        #pragma unroll
        for (int g = 0; g < 4; ++g) acc[mg][g] = (v8f){};

    #pragma unroll 2
    for (int k0 = 0; k0 < kH; k0 += 32) {
        union { v16bf v; v8bf h[2]; } a0, a1;
        a0.h[0] = *(const v8bf*)(aRow0 + k0);
        a0.h[1] = *(const v8bf*)(aRow0 + k0 + 8);
        a1.h[0] = *(const v8bf*)(aRow1 + k0);
        a1.h[1] = *(const v8bf*)(aRow1 + k0 + 8);

        const v16bf bG = *(const v16bf*)(bg_p + k0);
        const v16bf bI = *(const v16bf*)(bi_p + k0);
        const v16bf bF = *(const v16bf*)(bf_p + k0);
        const v16bf bO = *(const v16bf*)(bo_p + k0);

        // Each B fragment feeds two WMMAs (both M-subtiles).
        acc[0][0] = __builtin_amdgcn_wmma_f32_16x16x32_bf16(false, a0.v, false, bG,
                                                            (short)0, acc[0][0], false, false);
        acc[1][0] = __builtin_amdgcn_wmma_f32_16x16x32_bf16(false, a1.v, false, bG,
                                                            (short)0, acc[1][0], false, false);
        acc[0][1] = __builtin_amdgcn_wmma_f32_16x16x32_bf16(false, a0.v, false, bI,
                                                            (short)0, acc[0][1], false, false);
        acc[1][1] = __builtin_amdgcn_wmma_f32_16x16x32_bf16(false, a1.v, false, bI,
                                                            (short)0, acc[1][1], false, false);
        acc[0][2] = __builtin_amdgcn_wmma_f32_16x16x32_bf16(false, a0.v, false, bF,
                                                            (short)0, acc[0][2], false, false);
        acc[1][2] = __builtin_amdgcn_wmma_f32_16x16x32_bf16(false, a1.v, false, bF,
                                                            (short)0, acc[1][2], false, false);
        acc[0][3] = __builtin_amdgcn_wmma_f32_16x16x32_bf16(false, a0.v, false, bO,
                                                            (short)0, acc[0][3], false, false);
        acc[1][3] = __builtin_amdgcn_wmma_f32_16x16x32_bf16(false, a1.v, false, bO,
                                                            (short)0, acc[1][3], false, false);
    }

    // Epilogue. C/D layout: VGPR r, lanes 0-15 -> M=r, lanes 16-31 -> M=8+r;
    // N = lane%16. Gate combine is pure register math per lane.
    const int col = n0 + lmod;
    const float gxw = fgx[col];
    const float ixw = fixv[col];
    const float fxw = ffx[col];
    const float oxw = foxv[col];
    const float bgc = bg[col];

    #pragma unroll
    for (int mg = 0; mg < 2; ++mg) {
        const int rbase = m0 + mg * 16 + lhi * 8;
        #pragma unroll
        for (int r = 0; r < 8; ++r) {
            const int row = rbase + r;
            const float xv = x[(size_t)row * kS + t];   // x-projection (input dim 1)
            const size_t idx = (size_t)row * kH + col;
            const float g = tanhf(acc[mg][0][r] + xv * gxw + bgc);
            const float i = sigmoidf_fast(acc[mg][1][r] + xv * ixw + bgc);
            const float f = sigmoidf_fast(acc[mg][2][r] + xv * fxw + bgc);
            const float o = sigmoidf_fast(acc[mg][3][r] + xv * oxw + bgc);
            const float cn = fmaf(g, i, cbuf[idx] * f);
            cbuf[idx] = cn;
            const float hn = tanhf(cn) * o;
            hf32[idx] = hn;
            hout[idx] = (__bf16)hn;
        }
    }
}

// ---------------------------------------------------------------------------
// Head: p = h @ fc_ph.T + b_p, softmax over C=10. One block per batch row.
// C=10 is far too small for WMMA; plain VALU reduction (~5 MFLOP total).
// ---------------------------------------------------------------------------
__global__ __launch_bounds__(256)
void head_softmax(const float* __restrict__ h, const float* __restrict__ ph,
                  const float* __restrict__ bp, float* __restrict__ out) {
    const int b = blockIdx.x;
    const int tid = threadIdx.x;
    __shared__ float red[kC][256];

    float p[kC];
    #pragma unroll
    for (int c = 0; c < kC; ++c) p[c] = 0.0f;

    for (int k = tid; k < kH; k += 256) {
        const float hv = h[(size_t)b * kH + k];
        #pragma unroll
        for (int c = 0; c < kC; ++c) p[c] = fmaf(hv, ph[(size_t)c * kH + k], p[c]);
    }
    #pragma unroll
    for (int c = 0; c < kC; ++c) red[c][tid] = p[c];
    __syncthreads();

    for (int s = 128; s > 0; s >>= 1) {
        if (tid < s) {
            #pragma unroll
            for (int c = 0; c < kC; ++c) red[c][tid] += red[c][tid + s];
        }
        __syncthreads();
    }

    if (tid == 0) {
        float v[kC];
        float mx = -3.0e38f;
        #pragma unroll
        for (int c = 0; c < kC; ++c) { v[c] = red[c][0] + bp[c]; mx = fmaxf(mx, v[c]); }
        float sum = 0.0f;
        #pragma unroll
        for (int c = 0; c < kC; ++c) { v[c] = __expf(v[c] - mx); sum += v[c]; }
        const float inv = 1.0f / sum;
        #pragma unroll
        for (int c = 0; c < kC; ++c) out[(size_t)b * kC + c] = v[c] * inv;
    }
}

// ---------------------------------------------------------------------------
extern "C" void kernel_launch(void* const* d_in, const int* in_sizes, int n_in,
                              void* d_out, int out_size, void* d_ws, size_t ws_size,
                              hipStream_t stream) {
    (void)in_sizes; (void)n_in; (void)out_size; (void)ws_size;

    const float* x    = (const float*)d_in[0];
    const float* fgx  = (const float*)d_in[1];
    const float* Wg32 = (const float*)d_in[2];
    const float* bg   = (const float*)d_in[3];
    const float* fixv = (const float*)d_in[4];
    const float* Wi32 = (const float*)d_in[5];
    // d_in[6] = b_i : unused (reference applies b_g to all gates)
    const float* ffx  = (const float*)d_in[7];
    const float* Wf32 = (const float*)d_in[8];
    // d_in[9] = b_f : unused
    const float* foxv = (const float*)d_in[10];
    const float* Wo32 = (const float*)d_in[11];
    // d_in[12] = b_o : unused
    const float* ph   = (const float*)d_in[13];
    const float* bp   = (const float*)d_in[14];

    // Workspace layout (all offsets 256B-aligned):
    char* ws = (char*)d_ws;
    size_t off = 0;
    const size_t wBytes = (size_t)kH * kH * sizeof(__bf16);  // 2 MB each
    __bf16* Wg = (__bf16*)(ws + off); off += wBytes;
    __bf16* Wi = (__bf16*)(ws + off); off += wBytes;
    __bf16* Wf = (__bf16*)(ws + off); off += wBytes;
    __bf16* Wo = (__bf16*)(ws + off); off += wBytes;
    const size_t hbBytes = (size_t)kB * kH * sizeof(__bf16); // 512 KB
    __bf16* hA = (__bf16*)(ws + off); off += hbBytes;
    __bf16* hB = (__bf16*)(ws + off); off += hbBytes;
    const size_t hfBytes = (size_t)kB * kH * sizeof(float);  // 1 MB
    float* cbuf = (float*)(ws + off); off += hfBytes;
    float* hf32 = (float*)(ws + off); off += hfBytes;        // total ~11 MB

    const int nW = kH * kH;
    convert_weights<<<(nW + 255) / 256, 256, 0, stream>>>(
        Wg32, Wi32, Wf32, Wo32, Wg, Wi, Wf, Wo, nW);

    const int nS = kB * kH;
    init_state<<<(nS + 255) / 256, 256, 0, stream>>>(hA, cbuf, nS);

    const dim3 grid(kB / 32, kH / 64);   // 8 x 16 = 128 workgroups, 4 waves each
    for (int t = 0; t < kS; ++t) {
        const __bf16* hin = (t & 1) ? hB : hA;
        __bf16*       ho  = (t & 1) ? hA : hB;
        lstm_step<<<grid, 128, 0, stream>>>(hin, ho, cbuf, hf32,
                                            Wg, Wi, Wf, Wo,
                                            x, fgx, fixv, ffx, foxv, bg, t);
    }

    head_softmax<<<kB, 256, 0, stream>>>(hf32, ph, bp, (float*)d_out);
}
